// LinearAttention_38671885533892
// MI455X (gfx1250) — matrix operand order repaired and checked by reading, not compiled, not run on hardware
//
#include <hip/hip_runtime.h>
#include <hip/hip_bf16.h>

// ---------------------------------------------------------------------------
// LinearAttention (SAGAN-style) for MI455X / gfx1250.
// wave32, v_wmma_f32_16x16x32_bf16, TDM tensor_load_to_lds staging (when the
// builtin exists), double-buffered LDS, transposed operand layouts so every
// GEMM stages contiguous 32-element bf16 rows.
// ---------------------------------------------------------------------------

typedef __attribute__((ext_vector_type(16))) __bf16 v16bf;
typedef __attribute__((ext_vector_type(8)))  __bf16 v8bf;
typedef __attribute__((ext_vector_type(8)))  float  v8f;
typedef __attribute__((ext_vector_type(4)))  unsigned int v4u;
typedef __attribute__((ext_vector_type(4)))  int    v4i;
typedef __attribute__((ext_vector_type(8)))  int    v8i;

#define NB     16
#define CIN    256
#define CKEY   512
#define CVAL   512
#define NHEADS 8
#define DK     64
#define DV     64
#define S_SP   4096

#ifndef __has_builtin
#define __has_builtin(x) 0
#endif
#if defined(__gfx1250__) && __has_builtin(__builtin_amdgcn_tensor_load_to_lds)
#define HAVE_TDM 1
#else
#define HAVE_TDM 0
#endif

// ---------------------------------------------------------------------------
// Spectral-norm sigma: inv_sigma = 1 / || W @ normalize(W^T u) ||
// ---------------------------------------------------------------------------
__global__ __launch_bounds__(256) void sn_sigma_kernel(
    const float* __restrict__ W, const float* __restrict__ u,
    int O, int C, float* __restrict__ inv_sigma)
{
  __shared__ float sv[512];
  __shared__ float red[256];
  const int tid = threadIdx.x;

  for (int c = tid; c < C; c += 256) {
    float acc = 0.f;
    for (int o = 0; o < O; ++o) acc = fmaf(W[(size_t)o * C + c], u[o], acc);
    sv[c] = acc;
  }
  __syncthreads();

  float loc = 0.f;
  for (int c = tid; c < C; c += 256) loc += sv[c] * sv[c];
  red[tid] = loc; __syncthreads();
  for (int off = 128; off > 0; off >>= 1) {
    if (tid < off) red[tid] += red[tid + off];
    __syncthreads();
  }
  float invn = 1.f / fmaxf(sqrtf(red[0]), 1e-12f);
  __syncthreads();
  for (int c = tid; c < C; c += 256) sv[c] *= invn;
  __syncthreads();

  float s2 = 0.f;
  for (int o = tid; o < O; o += 256) {
    float acc = 0.f;
    for (int c = 0; c < C; ++c) acc = fmaf(W[(size_t)o * C + c], sv[c], acc);
    s2 += acc * acc;
  }
  red[tid] = s2; __syncthreads();
  for (int off = 128; off > 0; off >>= 1) {
    if (tid < off) red[tid] += red[tid + off];
    __syncthreads();
  }
  if (tid == 0) inv_sigma[0] = 1.f / fmaxf(sqrtf(red[0]), 1e-20f);
}

// ---------------------------------------------------------------------------
// fp32 -> bf16 with optional 1/sigma scale
// ---------------------------------------------------------------------------
__global__ __launch_bounds__(256) void scale_to_bf16_kernel(
    const float* __restrict__ in, __bf16* __restrict__ out,
    const float* __restrict__ inv_sigma, int n)
{
  int i = blockIdx.x * 256 + threadIdx.x;
  if (i >= n) return;
  float s = inv_sigma ? inv_sigma[0] : 1.f;
  out[i] = (__bf16)(in[i] * s);
}

// ---------------------------------------------------------------------------
// x [N][C][S] fp32 -> xt [N][S][C] bf16 (32x32 LDS tile transpose)
// ---------------------------------------------------------------------------
__global__ __launch_bounds__(256) void transpose_to_bf16_kernel(
    const float* __restrict__ in, __bf16* __restrict__ out)
{
  __shared__ float t[32][33];
  const int n  = blockIdx.z;
  const int s0 = blockIdx.x * 32;
  const int c0 = blockIdx.y * 32;
  const int tx = threadIdx.x & 31;
  const int ty = threadIdx.x >> 5;    // 32 x 8
  const float* ip = in  + (size_t)n * CIN  * S_SP;
  __bf16*      op = out + (size_t)n * S_SP * CIN;
  for (int i = 0; i < 32; i += 8)
    t[ty + i][tx] = ip[(size_t)(c0 + ty + i) * S_SP + s0 + tx];
  __syncthreads();
  for (int i = 0; i < 32; i += 8)
    op[(size_t)(s0 + ty + i) * CIN + c0 + tx] = (__bf16)t[tx][ty + i];
}

// ---------------------------------------------------------------------------
// TDM 2D tile load: ROWS rows of 32 bf16 (64 B), row stride ld elements,
// into LDS with 32 B padding after each 64 B row (96 B LDS row = 48 elems).
// ---------------------------------------------------------------------------
#if HAVE_TDM
__device__ __forceinline__ void tdm_load_tile(const __bf16* gsrc, unsigned int lds_off,
                                              int rows, int ldelems)
{
  unsigned long long ga = (unsigned long long)(uintptr_t)gsrc;
  v4u g0;
  g0.x = 1u;                                        // count=1, user desc
  g0.y = lds_off;                                   // lds_addr (bytes)
  g0.z = (unsigned int)(ga & 0xffffffffu);          // global_addr lo
  g0.w = (unsigned int)((ga >> 32) & 0x01ffffffu) | 0x80000000u;  // hi | type=2
  v8i g1;
  g1[0] = (1 << 16)      // data_size = 2 bytes
        | (1 << 20)      // pad_enable
        | (3 << 22)      // pad_interval: every 16 DWORDs (64 B)
        | (7 << 25);     // pad_amount: 8 DWORDs (32 B)
  g1[1] = 32 << 16;                            // tensor_dim0 = 32 (lo16)
  g1[2] = (rows & 0xffff) << 16;               // tensor_dim0 hi | tensor_dim1 lo
  g1[3] = 32 << 16;                            // tensor_dim1 hi | tile_dim0 = 32
  g1[4] = rows & 0xffff;                       // tile_dim1 = rows, tile_dim2 = 0
  g1[5] = ldelems;                             // tensor_dim0_stride lo32
  g1[6] = 0;
  g1[7] = 0;
  v4i z4 = {0, 0, 0, 0};
#if __clang_major__ >= 23
  v8i z8 = {0, 0, 0, 0, 0, 0, 0, 0};
  __builtin_amdgcn_tensor_load_to_lds(g0, g1, z4, z4, z8, 0);
#else
  __builtin_amdgcn_tensor_load_to_lds(g0, g1, z4, z4, 0);
#endif
}
#endif

template<int ROWS>
__device__ __forceinline__ void stage_fallback(const __bf16* __restrict__ g,
                                               __bf16* l, int ld, int tid)
{
  for (int e = tid; e < ROWS * 4; e += 256) {
    int r = e >> 2, c = (e & 3) * 8;
    *(v8bf*)&l[r * 48 + c] = *(const v8bf*)&g[(size_t)r * ld + c];
  }
}

// ---------------------------------------------------------------------------
// Tiled bf16 GEMM, C(MxN) = A(MxK, row-major, lda) * B, where B is stored
// N x K row-major (reduction dim contiguous, ldb = row stride in elements).
// 256 threads = 8 waves; per-wave tile (16*WMF) x (16*WNF); BK = 32.
// Double-buffered LDS staged by TDM (or vectorized copies).
// Epilogues: 0 bf16 acc+bias[row] | 1 bf16 acc | 2 bf16 transposed C[col][row]
//            3 fp32 gamma*(acc+bias)+resid      | 4 like 2 with +bias[row]
// C batch offset: (b>>bShift)*sC + (b&bMask)*sC2
// ---------------------------------------------------------------------------
template<int BM, int BN, int WMF, int WNF, int EPI>
__global__ __launch_bounds__(256) void gemm_bt_kernel(
    const __bf16* __restrict__ A, const __bf16* __restrict__ B,
    void* __restrict__ Cp, int K, int lda, int ldb, int ldc,
    long sA, long sB, long sC, long sC2, int bShift, int bMask,
    const float* __restrict__ bias, const float* __restrict__ gammap,
    const float* __restrict__ resid, long sR)
{
  constexpr int BK  = 32;
  constexpr int SAp = 48;                 // 96 B padded LDS row

  __shared__ __align__(32) __bf16 As[2 * BM * SAp];
  __shared__ __align__(32) __bf16 Bs[2 * BN * SAp];

  const int tid = threadIdx.x;
  const int b   = blockIdx.z;
  const __bf16* Ab = A + (size_t)b * sA;
  const __bf16* Bb = B + (size_t)b * sB;
  const int m0 = blockIdx.y * BM;
  const int n0 = blockIdx.x * BN;

  const int wave     = tid >> 5;
  const int lane     = tid & 31;
  const int laneHalf = lane >> 4;
  const int lane16   = lane & 15;
  const int wm0 = (wave & 3) * 16 * WMF;   // 4 wave rows
  const int wn0 = (wave >> 2) * 16 * WNF;  // 2 wave cols

  v8f acc[WMF][WNF];
  v8f zero = {0.f, 0.f, 0.f, 0.f, 0.f, 0.f, 0.f, 0.f};
  for (int i = 0; i < WMF; ++i)
    for (int j = 0; j < WNF; ++j) acc[i][j] = zero;

  const int nk = K / BK;

  // ---- stage one k-step into buffer `buf` ----
  auto issue = [&](int kk, int buf) {
    const __bf16* gA = Ab + (size_t)m0 * lda + kk * BK;
    const __bf16* gB = Bb + (size_t)n0 * ldb + kk * BK;
    __bf16* lA = As + buf * (BM * SAp);
    __bf16* lB = Bs + buf * (BN * SAp);
#if HAVE_TDM
    if (wave == 0)      tdm_load_tile(gA, (unsigned int)(uintptr_t)lA, BM, lda);
    else if (wave == 1) tdm_load_tile(gB, (unsigned int)(uintptr_t)lB, BN, ldb);
#else
    stage_fallback<BM>(gA, lA, lda, tid);
    stage_fallback<BN>(gB, lB, ldb, tid);
#endif
  };

  issue(0, 0);
  for (int kk = 0; kk < nk; ++kk) {
#if HAVE_TDM
    if (wave < 2) __builtin_amdgcn_s_wait_tensorcnt(0);
#endif
    __syncthreads();   // buf kk&1 ready for all; all reads of buf (kk+1)&1 done
    if (kk + 1 < nk) issue(kk + 1, (kk + 1) & 1);

    const __bf16* lA = As + (kk & 1) * (BM * SAp);
    const __bf16* lB = Bs + (kk & 1) * (BN * SAp);

    v16bf af[WMF], bfr[WNF];
    for (int i = 0; i < WMF; ++i) {
      // A 16x32 bf16 frag: lanes 0-15 -> K{0..7,16..23}, lanes 16-31 -> K{8..15,24..31}
      const __bf16* p = &lA[(wm0 + i * 16 + lane16) * SAp];
      union { v16bf v; v8bf h[2]; } u;
      u.h[0] = *(const v8bf*)(p + laneHalf * 8);
      u.h[1] = *(const v8bf*)(p + 16 + laneHalf * 8);
      af[i] = u.v;
    }
    for (int j = 0; j < WNF; ++j) {
      // B 32x16 frag: lanes 0-15 -> K{0..7,16..23}, lanes 16-31 -> K{8..15,24..31}, col = lane&15
      const __bf16* p = &lB[(wn0 + j * 16 + lane16) * SAp];
      union { v16bf v; v8bf h[2]; } u;
      u.h[0] = *(const v8bf*)(p + laneHalf * 8);
      u.h[1] = *(const v8bf*)(p + 16 + laneHalf * 8);
      bfr[j] = u.v;
    }
    for (int i = 0; i < WMF; ++i)
      for (int j = 0; j < WNF; ++j)
        acc[i][j] = __builtin_amdgcn_wmma_f32_16x16x32_bf16(
            false, af[i], false, bfr[j], (short)0, acc[i][j], false, false);
  }

  // ---- epilogue (C frag: row = laneHalf*8 + g, col = lane&15) ----
  const size_t offC = (size_t)(b >> bShift) * (size_t)sC + (size_t)(b & bMask) * (size_t)sC2;
  float gm = (EPI == 3) ? gammap[0] : 0.f;
  for (int i = 0; i < WMF; ++i) {
    for (int j = 0; j < WNF; ++j) {
      int rbase = m0 + wm0 + i * 16 + laneHalf * 8;
      int col   = n0 + wn0 + j * 16 + lane16;
      if (EPI == 2 || EPI == 4) {
        v8bf pk;
        for (int g = 0; g < 8; ++g) {
          float v = acc[i][j][g];
          if (EPI == 4) v += bias[rbase + g];
          pk[g] = (__bf16)v;
        }
        *(v8bf*)&((__bf16*)Cp)[offC + (size_t)col * ldc + rbase] = pk;
      } else {
        for (int g = 0; g < 8; ++g) {
          int row = rbase + g;
          float v = acc[i][j][g];
          if (EPI == 0) {
            ((__bf16*)Cp)[offC + (size_t)row * ldc + col] = (__bf16)(v + bias[row]);
          } else if (EPI == 1) {
            ((__bf16*)Cp)[offC + (size_t)row * ldc + col] = (__bf16)v;
          } else {
            size_t idx = (size_t)row * ldc + col;
            ((float*)Cp)[offC + idx] =
                gm * (v + bias[row]) + resid[(size_t)b * sR + idx];
          }
        }
      }
    }
  }
}

// ---------------------------------------------------------------------------
// Row softmax over s (4096), one block per row, bf16 in/out, vectorized.
// ---------------------------------------------------------------------------
__global__ __launch_bounds__(256) void softmax_rows_kernel(
    const __bf16* __restrict__ in, __bf16* __restrict__ out)
{
  __shared__ float red[256];
  const int tid = threadIdx.x;
  const __bf16* rp = in  + (size_t)blockIdx.x * S_SP + tid * 16;
  __bf16*       op = out + (size_t)blockIdx.x * S_SP + tid * 16;

  v8bf r0 = *(const v8bf*)rp;
  v8bf r1 = *(const v8bf*)(rp + 8);
  float vals[16];
  float mx = -3.4e38f;
  for (int i = 0; i < 8; ++i) { vals[i] = (float)r0[i];     mx = fmaxf(mx, vals[i]); }
  for (int i = 0; i < 8; ++i) { vals[8 + i] = (float)r1[i]; mx = fmaxf(mx, vals[8 + i]); }
  red[tid] = mx; __syncthreads();
  for (int off = 128; off > 0; off >>= 1) {
    if (tid < off) red[tid] = fmaxf(red[tid], red[tid + off]);
    __syncthreads();
  }
  mx = red[0]; __syncthreads();

  float sum = 0.f;
  for (int i = 0; i < 16; ++i) { vals[i] = __expf(vals[i] - mx); sum += vals[i]; }
  red[tid] = sum; __syncthreads();
  for (int off = 128; off > 0; off >>= 1) {
    if (tid < off) red[tid] += red[tid + off];
    __syncthreads();
  }
  float inv = 1.f / red[0];
  v8bf w0, w1;
  for (int i = 0; i < 8; ++i) { w0[i] = (__bf16)(vals[i] * inv); w1[i] = (__bf16)(vals[8 + i] * inv); }
  *(v8bf*)op = w0;
  *(v8bf*)(op + 8) = w1;
}

// ---------------------------------------------------------------------------
// Column softmax over 64 key channels per head.
// in  = qbT [n][s][512]  (channel-contiguous), out = qst [n*8+h][s][64]
// ---------------------------------------------------------------------------
__global__ __launch_bounds__(256) void softmax_qcols_kernel(
    const __bf16* __restrict__ in, __bf16* __restrict__ out)
{
  size_t gt = (size_t)blockIdx.x * 256 + threadIdx.x;   // [0, 128*4096)
  size_t bh = gt >> 12;
  size_t s  = gt & 4095;
  const __bf16* p = in  + (bh >> 3) * (size_t)S_SP * CKEY + s * CKEY + (bh & 7) * 64;
  __bf16*       o = out + bh * (size_t)S_SP * 64 + s * 64;

  float v[64];
  float mx = -3.4e38f;
#pragma unroll
  for (int c = 0; c < 8; ++c) {
    v8bf x = *(const v8bf*)(p + c * 8);
#pragma unroll
    for (int e = 0; e < 8; ++e) { v[c * 8 + e] = (float)x[e]; mx = fmaxf(mx, v[c * 8 + e]); }
  }
  float sum = 0.f;
#pragma unroll
  for (int j = 0; j < 64; ++j) { v[j] = __expf(v[j] - mx); sum += v[j]; }
  float inv = 1.f / sum;
#pragma unroll
  for (int c = 0; c < 8; ++c) {
    v8bf w;
#pragma unroll
    for (int e = 0; e < 8; ++e) w[e] = (__bf16)(v[c * 8 + e] * inv);
    *(v8bf*)(o + c * 8) = w;
  }
}

// ---------------------------------------------------------------------------
extern "C" void kernel_launch(void* const* d_in, const int* in_sizes, int n_in,
                              void* d_out, int out_size, void* d_ws, size_t ws_size,
                              hipStream_t stream)
{
  const float* input = (const float*)d_in[0];
  const float* Wk = (const float*)d_in[1];  const float* bk = (const float*)d_in[2];
  const float* uk = (const float*)d_in[3];
  const float* Wq = (const float*)d_in[4];  const float* bq = (const float*)d_in[5];
  const float* uq = (const float*)d_in[6];
  const float* Wv = (const float*)d_in[7];  const float* bv = (const float*)d_in[8];
  const float* uv = (const float*)d_in[9];
  const float* Wr = (const float*)d_in[10]; const float* br = (const float*)d_in[11];
  const float* ur = (const float*)d_in[12];
  const float* gamma = (const float*)d_in[13];
  float* out = (float*)d_out;

  char* ws = (char*)d_ws;
  size_t off = 0;
  auto take = [&](size_t bytes) -> char* {
    char* p = ws + off;
    off += (bytes + 255) & ~(size_t)255;
    return p;
  };

  __bf16* xt   = (__bf16*)take((size_t)NB * S_SP * CIN * 2);   // [n][s][c]
  __bf16* kb   = (__bf16*)take((size_t)NB * CKEY * S_SP * 2);  // [n][o][s]
  __bf16* qbT  = (__bf16*)take((size_t)NB * S_SP * CKEY * 2);  // [n][s][o]
  __bf16* vb   = (__bf16*)take((size_t)NB * CVAL * S_SP * 2);  // [n][o][s]
  __bf16* ks   = (__bf16*)take((size_t)NB * CKEY * S_SP * 2);  // [n][o][s]
  __bf16* qst  = (__bf16*)take((size_t)NB * NHEADS * S_SP * DK * 2);  // [bh][s][64]
  __bf16* attT = (__bf16*)take((size_t)NB * S_SP * CVAL * 2);  // [n][s][512]
  __bf16* ctxT = (__bf16*)take((size_t)NB * NHEADS * DK * DV * 2);    // [bh][v][k]
  __bf16* Wkb  = (__bf16*)take((size_t)CKEY * CIN * 2);
  __bf16* Wqb  = (__bf16*)take((size_t)CKEY * CIN * 2);
  __bf16* Wvb  = (__bf16*)take((size_t)CVAL * CIN * 2);
  __bf16* Wrb  = (__bf16*)take((size_t)CIN * CVAL * 2);
  float* sigk = (float*)take(256);
  float* sigq = (float*)take(256);
  float* sigv = (float*)take(256);
  float* sigr = (float*)take(256);

  // ---- spectral-norm sigmas ----
  sn_sigma_kernel<<<1, 256, 0, stream>>>(Wk, uk, CKEY, CIN, sigk);
  sn_sigma_kernel<<<1, 256, 0, stream>>>(Wq, uq, CKEY, CIN, sigq);
  sn_sigma_kernel<<<1, 256, 0, stream>>>(Wv, uv, CVAL, CIN, sigv);
  sn_sigma_kernel<<<1, 256, 0, stream>>>(Wr, ur, CIN, CVAL, sigr);

  // ---- weights -> bf16 * (1/sigma) ----
  {
    int nW = CKEY * CIN;
    scale_to_bf16_kernel<<<(nW + 255) / 256, 256, 0, stream>>>(Wk, Wkb, sigk, nW);
    scale_to_bf16_kernel<<<(nW + 255) / 256, 256, 0, stream>>>(Wq, Wqb, sigq, nW);
    scale_to_bf16_kernel<<<(nW + 255) / 256, 256, 0, stream>>>(Wv, Wvb, sigv, nW);
    int nR = CIN * CVAL;
    scale_to_bf16_kernel<<<(nR + 255) / 256, 256, 0, stream>>>(Wr, Wrb, sigr, nR);
  }

  // ---- x -> xt (bf16, transposed [n][s][c]) ----
  {
    dim3 g(S_SP / 32, CIN / 32, NB);
    transpose_to_bf16_kernel<<<g, 256, 0, stream>>>(input, xt);
  }

  // ---- QKV: C[n] = Wb (512x256) @ xt[n]^T,  B = xt (s-major, k=c contiguous) ----
  {
    dim3 g(S_SP / 128, CKEY / 128, NB);
    gemm_bt_kernel<128, 128, 2, 4, 0><<<g, 256, 0, stream>>>(
        Wkb, xt, kb, CIN, CIN, CIN, S_SP,
        0L, (long)S_SP * CIN, (long)CKEY * S_SP, 0L, 0, 0,
        bk, nullptr, nullptr, 0L);
    gemm_bt_kernel<128, 128, 2, 4, 4><<<g, 256, 0, stream>>>(   // transposed out + bias
        Wqb, xt, qbT, CIN, CIN, CIN, CKEY,
        0L, (long)S_SP * CIN, (long)S_SP * CKEY, 0L, 0, 0,
        bq, nullptr, nullptr, 0L);
    gemm_bt_kernel<128, 128, 2, 4, 0><<<g, 256, 0, stream>>>(
        Wvb, xt, vb, CIN, CIN, CIN, S_SP,
        0L, (long)S_SP * CIN, (long)CVAL * S_SP, 0L, 0, 0,
        bv, nullptr, nullptr, 0L);
  }

  // ---- softmaxes ----
  softmax_rows_kernel<<<NB * CKEY, 256, 0, stream>>>(kb, ks);
  softmax_qcols_kernel<<<(NB * NHEADS * S_SP) / 256, 256, 0, stream>>>(qbT, qst);

  // ---- ctx[bh] (64x64) = Ks (64x4096) @ V^T ; stored transposed ctxT[v][k] ----
  {
    dim3 g(1, 1, NB * NHEADS);
    gemm_bt_kernel<64, 64, 1, 2, 2><<<g, 256, 0, stream>>>(
        ks, vb, ctxT, S_SP, S_SP, S_SP, DK,
        (long)DK * S_SP, (long)DV * S_SP, (long)DK * DV, 0L, 0, 0,
        nullptr, nullptr, nullptr, 0L);
  }

  // ---- att[bh] (64 x 4096) = ctxT (64x64) @ Qs ; stored as attT [n][s][512] ----
  {
    dim3 g(S_SP / 64, 1, NB * NHEADS);
    gemm_bt_kernel<64, 64, 1, 2, 2><<<g, 256, 0, stream>>>(
        ctxT, qst, attT, DK, DK, DK, CKEY,
        (long)DK * DV, (long)S_SP * DK, (long)S_SP * CKEY, 64L, 3, 7,
        nullptr, nullptr, nullptr, 0L);
  }

  // ---- out[n] = gamma * (Wrb (256x512) @ att[n] + br) + input[n] ----
  {
    dim3 g(S_SP / 128, CIN / 128, NB);
    gemm_bt_kernel<128, 128, 2, 4, 3><<<g, 256, 0, stream>>>(
        Wrb, attT, out, CVAL, CVAL, CVAL, S_SP,
        0L, (long)S_SP * CVAL, (long)CIN * S_SP, 0L, 0, 0,
        br, gamma, input, (long)CIN * S_SP);
  }
}
